// ConvOnlyNet_83708912599759
// MI455X (gfx1250) — matrix operand order repaired and checked by reading, not compiled, not run on hardware
//
#include <hip/hip_runtime.h>
#include <cstdint>

typedef __attribute__((ext_vector_type(16))) _Float16 v16h;
typedef __attribute__((ext_vector_type(8)))  float    v8f;
typedef __attribute__((ext_vector_type(4)))  unsigned int u32x4;

union F16Frag { u32x4 u[2]; v16h h; };
union PK8  { _Float16 f[8];  u32x4 u;    };
union PK32 { _Float16 f[32]; u32x4 u[4]; };

__device__ __forceinline__ v8f wmma_f16(const F16Frag& a, const F16Frag& b, v8f c) {
  // D = A(16x32 f16) * B(32x16 f16) + C(16x16 f32)
  return __builtin_amdgcn_wmma_f32_16x16x32_f16(false, a.h, false, b.h, (short)0, c,
                                                false, false);
}

// Async global->LDS copy (16B per lane), tracked on ASYNCcnt. LDS destination
// address is the low 32 bits of the flat shared pointer (LDS aperture lives in
// the high bits on gfx1250).
__device__ __forceinline__ void async_copy_b128(unsigned lds_off, const void* g) {
  asm volatile("global_load_async_to_lds_b128 %0, %1, off"
               :
               : "v"(lds_off), "v"((unsigned long long)(uintptr_t)g)
               : "memory");
}
__device__ __forceinline__ void async_wait0() {
  asm volatile("s_wait_asynccnt 0x0" ::: "memory");
}
__device__ __forceinline__ unsigned lds_addr_of(const void* p) {
  return (unsigned)(uintptr_t)p;
}

// ---------------------------------------------------------------------------
// Weight repack: fp32 [OC][IC][3][3] -> fp16 [OCp][9][IC] (tap-major, ic
// contiguous) so WMMA A-fragments are contiguous 16B chunks. oc >= OC padded
// with zeros.
// ---------------------------------------------------------------------------
__global__ __launch_bounds__(256) void repack_w(const float* __restrict__ w,
                                                _Float16* __restrict__ wpk,
                                                int OCp, int OC, int IC) {
  int i = blockIdx.x * 256 + threadIdx.x;
  int total = OCp * 9 * IC;
  if (i >= total) return;
  int ic = i % IC;
  int t  = (i / IC) % 9;
  int oc = i / (IC * 9);
  float v = (oc < OC) ? w[((size_t)oc * IC + ic) * 9 + t] : 0.f;
  wpk[i] = (_Float16)v;
}

// ---------------------------------------------------------------------------
// Zero the 1-pixel border of a padded channels-last fp16 buffer [N][Hp][Wp][C].
// ---------------------------------------------------------------------------
__global__ __launch_bounds__(256) void zero_border(_Float16* __restrict__ buf,
                                                   int Hp, int Wp, int C, int total) {
  int idx = blockIdx.x * 256 + threadIdx.x;
  if (idx >= total) return;
  const int perim = 2 * Wp + 2 * (Hp - 2);
  int c = idx % C;
  int p = (idx / C) % perim;
  int n = idx / (C * perim);
  int y, x;
  if (p < Wp)            { y = 0;      x = p; }
  else if (p < 2 * Wp)   { y = Hp - 1; x = p - Wp; }
  else { int q = p - 2 * Wp; y = 1 + (q >> 1); x = (q & 1) ? (Wp - 1) : 0; }
  buf[((size_t)(n * Hp + y) * Wp + x) * C + c] = (_Float16)0.f;
}

// ---------------------------------------------------------------------------
// Layer 1: conv3x3 (1->32) + relu + 2x2 pool, direct VALU (K=9 too shallow
// for WMMA). Output: fp16 channels-last padded [64][130][130][32], packed
// 16B stores.
// ---------------------------------------------------------------------------
__global__ __launch_bounds__(256) void conv1_pool(const float* __restrict__ x,
                                                  const float* __restrict__ w,
                                                  const float* __restrict__ b,
                                                  _Float16* __restrict__ out) {
  __shared__ float sw[288];
  __shared__ float sb[32];
  const int tid = threadIdx.x;
  for (int e = tid; e < 288; e += 256) sw[e] = w[e];
  if (tid < 32) sb[tid] = b[tid];
  __syncthreads();

  const int idx = blockIdx.x * 256 + tid;     // over 64*128*128
  const int px = idx & 127;
  const int py = (idx >> 7) & 127;
  const int n  = idx >> 14;

  float patch[4][4];
#pragma unroll
  for (int dy = 0; dy < 4; ++dy)
#pragma unroll
    for (int dx = 0; dx < 4; ++dx) {
      const int yy = 2 * py - 1 + dy, xx = 2 * px - 1 + dx;
      patch[dy][dx] = (yy >= 0 && yy < 256 && xx >= 0 && xx < 256)
                          ? x[((size_t)n * 256 + yy) * 256 + xx] : 0.f;
    }

  PK32 pk;
  for (int oc = 0; oc < 32; ++oc) {
    float m = 0.f;  // relu => every pooled candidate >= 0
#pragma unroll
    for (int q = 0; q < 4; ++q) {
      const int dy = q >> 1, dx = q & 1;
      float a = sb[oc];
#pragma unroll
      for (int t = 0; t < 9; ++t)
        a += patch[dy + t / 3][dx + t % 3] * sw[oc * 9 + t];
      m = fmaxf(m, a);
    }
    pk.f[oc] = (_Float16)m;
  }
  _Float16* op = out + (((size_t)n * 130 + py + 1) * 130 + px + 1) * 32;
#pragma unroll
  for (int j = 0; j < 4; ++j) ((u32x4*)op)[j] = pk.u[j];
}

// ---------------------------------------------------------------------------
// WMMA implicit-GEMM conv3x3 + bias + relu + fused 2x2 maxpool.
// Input:  fp16 channels-last padded [N][H+2][W+2][IC] (zero border).
// Output: fp16 channels-last padded [N][H/2+2][W/2+2][OC].
// Staging: per-row contiguous global->LDS async b128 copies (ASYNCcnt).
// 8 waves = (OC/16 oc-tiles) x (CS col segments of 32 conv cols); each wave
// owns 4 accumulator tiles (2 rows x 2 col-halves of 16 px), pooled via LDS.
// ---------------------------------------------------------------------------
template <int IC, int OC, int H, int W, int CS>
__global__ __launch_bounds__(256) void conv_wmma_pool(
    const _Float16* __restrict__ in, const _Float16* __restrict__ wpk,
    const float* __restrict__ bias, _Float16* __restrict__ out) {
  constexpr int NOCT = OC / 16;
  static_assert(NOCT * CS == 8, "8 waves = oc-tiles x col-segments");
  constexpr int TILEW = 32 * CS;
  constexpr int SCOLS = TILEW + 2;
  constexpr int Hp = H + 2, Wp = W + 2;
  constexpr int KC = IC / 32;

  __shared__ _Float16 sIn[4][SCOLS][IC];
  __shared__ float sPool[8][16][2][32];   // [wave][m][conv row][conv col]

  const int wgPerRow = W / TILEW;
  const int n  = blockIdx.x / wgPerRow;
  const int xs = (blockIdx.x % wgPerRow) * TILEW;
  const int py = blockIdx.y;
  const int y0 = 2 * py;
  const int tid = threadIdx.x;

  // Async-stage conv rows y0-1..y0+2 (padded rows y0..y0+3), cols xs-1.. :
  // each staged row is one contiguous global chunk of SCOLS*IC halves.
  constexpr int ROWH   = SCOLS * IC;          // halves per staged row
  constexpr int NCHUNK = 4 * ROWH / 8;        // 16B chunks total
  const unsigned ldsBase = lds_addr_of(&sIn[0][0][0]);
  for (int e = tid; e < NCHUNK; e += 256) {
    const int row    = (e * 8) / ROWH;
    const int within = (e * 8) % ROWH;
    const _Float16* g = in + ((size_t)(n * Hp + y0 + row) * Wp + xs) * IC + within;
    async_copy_b128(ldsBase + e * 16, g);
  }
  async_wait0();
  __syncthreads();

  const int wave = tid >> 5;
  const int lane = tid & 31;
  const int g    = lane >> 4;
  const int nn   = lane & 15;
  const int oct  = wave % NOCT;
  const int cs   = wave / NOCT;
  const int oc0  = oct * 16;

  v8f acc[4] = {};

#pragma unroll
  for (int tap = 0; tap < 9; ++tap) {
    const int ky = tap / 3, kx = tap % 3;
#pragma unroll
    for (int kc = 0; kc < KC; ++kc) {
      F16Frag a;
      const _Float16* ap =
          wpk + ((size_t)(oc0 + nn) * 9 + tap) * IC + kc * 32 + 8 * g;
      a.u[0] = *(const u32x4*)ap;
      a.u[1] = *(const u32x4*)(ap + 16);
#pragma unroll
      for (int t = 0; t < 4; ++t) {
        const int r = t >> 1, c = t & 1;
        F16Frag bm;
        const _Float16* bp =
            &sIn[r + ky][cs * 32 + c * 16 + nn + kx][kc * 32 + 8 * g];
        bm.u[0] = *(const u32x4*)bp;
        bm.u[1] = *(const u32x4*)(bp + 16);
        acc[t] = wmma_f16(a, bm, acc[t]);
      }
    }
  }

  float bb[8];
#pragma unroll
  for (int v = 0; v < 8; ++v) bb[v] = bias[oc0 + v + 8 * g];

  // D layout: lane holds N = nn, VGPR v holds M = v + 8*g.
#pragma unroll
  for (int t = 0; t < 4; ++t) {
    const int r = t >> 1, c = t & 1;
#pragma unroll
    for (int v = 0; v < 8; ++v) {
      float val = fmaxf(acc[t][v] + bb[v], 0.f);
      sPool[wave][v + 8 * g][r][c * 16 + nn] = val;
    }
  }
  __syncthreads();

  // Pool 2x2 and store 8 consecutive oc (oc0+8g .. +7) as one 16B chunk.
  const int pxb = (xs + cs * 32) >> 1;
  PK8 pk;
#pragma unroll
  for (int v = 0; v < 8; ++v) {
    const int m = v + 8 * g;
    const float a0 = sPool[wave][m][0][2 * nn];
    const float a1 = sPool[wave][m][0][2 * nn + 1];
    const float a2 = sPool[wave][m][1][2 * nn];
    const float a3 = sPool[wave][m][1][2 * nn + 1];
    pk.f[v] = (_Float16)fmaxf(fmaxf(a0, a1), fmaxf(a2, a3));
  }
  constexpr int Hp2 = H / 2 + 2, Wp2 = W / 2 + 2;
  _Float16* op = out + ((size_t)(n * Hp2 + py + 1) * Wp2 + pxb + nn + 1) * OC +
                 oc0 + 8 * g;
  *(u32x4*)op = pk.u;
}

// ---------------------------------------------------------------------------
// Layer 4: WMMA conv3x3 (128 -> 10, oc padded to 16), + bias, no relu/pool.
// Input fp16 padded [64][34][34][128]; the staged 6-row block is one fully
// contiguous global region -> pure async b128 copies.
// ---------------------------------------------------------------------------
__global__ __launch_bounds__(256) void conv4_wmma(const _Float16* __restrict__ in,
                                                  const _Float16* __restrict__ wpk,
                                                  const float* __restrict__ bias,
                                                  float* __restrict__ out) {
  constexpr int IC = 128, H = 32, W = 32, Wp = 34;
  __shared__ _Float16 sIn[6][34][IC];

  const int n  = blockIdx.x;
  const int y0 = blockIdx.y * 4;
  const int tid = threadIdx.x;

  constexpr int NCHUNK = 6 * 34 * IC / 8;   // 16B chunks
  const unsigned ldsBase = lds_addr_of(&sIn[0][0][0]);
  const char* g0 = (const char*)(in + (size_t)(n * 34 + y0) * Wp * IC);
  for (int e = tid; e < NCHUNK; e += 256)
    async_copy_b128(ldsBase + e * 16, g0 + e * 16);
  async_wait0();
  __syncthreads();

  const int wave = tid >> 5;
  const int lane = tid & 31;
  const int g    = lane >> 4;
  const int nn   = lane & 15;
  const int r    = wave >> 1;
  const int c    = wave & 1;

  v8f acc = {};
#pragma unroll
  for (int tap = 0; tap < 9; ++tap) {
    const int ky = tap / 3, kx = tap % 3;
#pragma unroll
    for (int kc = 0; kc < IC / 32; ++kc) {
      F16Frag a;
      const _Float16* ap = wpk + ((size_t)nn * 9 + tap) * IC + kc * 32 + 8 * g;
      a.u[0] = *(const u32x4*)ap;
      a.u[1] = *(const u32x4*)(ap + 16);
      F16Frag bm;
      const _Float16* bp = &sIn[r + ky][c * 16 + nn + kx][kc * 32 + 8 * g];
      bm.u[0] = *(const u32x4*)bp;
      bm.u[1] = *(const u32x4*)(bp + 16);
      acc = wmma_f16(a, bm, acc);
    }
  }

  const int y = y0 + r;
  const int x = c * 16 + nn;
#pragma unroll
  for (int v = 0; v < 8; ++v) {
    const int m = v + 8 * g;
    if (m < 10)
      out[((size_t)(n * 10 + m) * H + y) * W + x] = acc[v] + bias[m];
  }
}

// ---------------------------------------------------------------------------
// NaN-aware log-softmax over rows of 10240. Any element < -1 (the 1x1-pool
// NaN threshold) poisons the entire row with NaN (max/logsumexp propagate).
// ---------------------------------------------------------------------------
__global__ __launch_bounds__(256) void logsoftmax_k(const float* __restrict__ in,
                                                    float* __restrict__ out) {
  __shared__ float red[256];
  const int n = blockIdx.x;
  const int tid = threadIdx.x;
  const float* row = in + (size_t)n * 10240;

  float lmax = -__builtin_inff();
  int ln = 0;
  for (int i = tid; i < 10240; i += 256) {
    const float v = row[i];
    if (v < -1.0f || v != v) ln = 1;
    lmax = fmaxf(lmax, v);
  }
  const int nanflag = __syncthreads_or(ln);

  red[tid] = lmax;
  __syncthreads();
  for (int s = 128; s > 0; s >>= 1) {
    if (tid < s) red[tid] = fmaxf(red[tid], red[tid + s]);
    __syncthreads();
  }
  const float gmax = red[0];
  __syncthreads();

  float ls = 0.f;
  for (int i = tid; i < 10240; i += 256) ls += expf(row[i] - gmax);
  red[tid] = ls;
  __syncthreads();
  for (int s = 128; s > 0; s >>= 1) {
    if (tid < s) red[tid] += red[tid + s];
    __syncthreads();
  }
  const float logZ = gmax + logf(red[0]);
  const float qnan = __int_as_float(0x7FC00000);

  for (int i = tid; i < 10240; i += 256)
    out[(size_t)n * 10240 + i] = nanflag ? qnan : (row[i] - logZ);
}

// ---------------------------------------------------------------------------
extern "C" void kernel_launch(void* const* d_in, const int* in_sizes, int n_in,
                              void* d_out, int out_size, void* d_ws, size_t ws_size,
                              hipStream_t stream) {
  (void)in_sizes; (void)n_in; (void)out_size; (void)ws_size;
  const float* x  = (const float*)d_in[0];
  const float* w1 = (const float*)d_in[1];
  const float* b1 = (const float*)d_in[2];
  const float* w2 = (const float*)d_in[3];
  const float* b2 = (const float*)d_in[4];
  const float* w3 = (const float*)d_in[5];
  const float* b3 = (const float*)d_in[6];
  const float* w4 = (const float*)d_in[7];
  const float* b4 = (const float*)d_in[8];
  float* out = (float*)d_out;

  char* ws = (char*)d_ws;
  size_t o = 0;
  _Float16* buf1 = (_Float16*)(ws + o); o += (size_t)64 * 130 * 130 * 32 * 2;  // L1 out, padded
  _Float16* buf2 = (_Float16*)(ws + o); o += (size_t)64 * 66 * 66 * 64 * 2;    // L2 out, padded
  _Float16* buf3 = (_Float16*)(ws + o); o += (size_t)64 * 34 * 34 * 128 * 2;   // L3 out, padded
  float*    buf4 = (float*)(ws + o);    o += (size_t)64 * 10 * 32 * 32 * 4;    // L4 out
  _Float16* wpk2 = (_Float16*)(ws + o); o += (size_t)64 * 9 * 32 * 2;
  _Float16* wpk3 = (_Float16*)(ws + o); o += (size_t)128 * 9 * 64 * 2;
  _Float16* wpk4 = (_Float16*)(ws + o); o += (size_t)16 * 9 * 128 * 2;

  repack_w<<<(64 * 9 * 32 + 255) / 256, 256, 0, stream>>>(w2, wpk2, 64, 64, 32);
  repack_w<<<(128 * 9 * 64 + 255) / 256, 256, 0, stream>>>(w3, wpk3, 128, 128, 64);
  repack_w<<<(16 * 9 * 128 + 255) / 256, 256, 0, stream>>>(w4, wpk4, 16, 10, 128);

  {  // zero the padded borders the async staging relies on
    int t1 = 64 * (2 * 130 + 2 * 128) * 32;
    int t2 = 64 * (2 * 66 + 2 * 64) * 64;
    int t3 = 64 * (2 * 34 + 2 * 32) * 128;
    zero_border<<<(t1 + 255) / 256, 256, 0, stream>>>(buf1, 130, 130, 32, t1);
    zero_border<<<(t2 + 255) / 256, 256, 0, stream>>>(buf2, 66, 66, 64, t2);
    zero_border<<<(t3 + 255) / 256, 256, 0, stream>>>(buf3, 34, 34, 128, t3);
  }

  conv1_pool<<<(64 * 128 * 128) / 256, 256, 0, stream>>>(x, w1, b1, buf1);

  conv_wmma_pool<32, 64, 128, 128, 2>
      <<<dim3(64 * (128 / 64), 64), 256, 0, stream>>>(buf1, wpk2, b2, buf2);

  conv_wmma_pool<64, 128, 64, 64, 1>
      <<<dim3(64 * (64 / 32), 32), 256, 0, stream>>>(buf2, wpk3, b3, buf3);

  conv4_wmma<<<dim3(64, 8), 256, 0, stream>>>(buf3, wpk4, b4, buf4);

  logsoftmax_k<<<64, 256, 0, stream>>>(buf4, out);
}
